// OuterProductSum_82806969466888
// MI455X (gfx1250) — matrix operand order repaired and checked by reading, not compiled, not run on hardware
//
#include <hip/hip_runtime.h>

// ---------------------------------------------------------------------------
// OuterProductSum for MI455X (gfx1250, wave32, WMMA)
//
// out[i,j,e] = edge[i,j,e] + sum_k tmp[i,e,k]*b[j,k] + b_final[e]
//   tmp[i,e,k] = sum_d a[i,d] * W_final[e, d*32+k]
//   [a|b] = LN(node) @ W_embed^T + b_embed
//
// Main GEMM (4.83 GFLOP) done with v_wmma_f32_16x16x32_bf16 (f32 accumulate).
// HBM floor = 604 MB (edge in + out out) ~= 26 us @ 23.3 TB/s -> bandwidth
// bound once matrix math runs on the WMMA pipe.
// ---------------------------------------------------------------------------

#define N_NODES   768
#define NODE_DIM  384
#define EDGE_DIM  128
#define INNER     32
#define LN_EPS    1e-5f

typedef __attribute__((ext_vector_type(16))) __bf16 v16bf;
typedef __attribute__((ext_vector_type(8)))  float  v8f;

// ---------------------------------------------------------------------------
// Kernel 1: per-row LayerNorm + embed + tmp = a x Wr  (one block per row i)
//   writes b_bf16[i][k]      (k = 0..31)
//   writes tmp_bf16[i][e][k] (e = 0..127, k = 0..31), row-major
// ---------------------------------------------------------------------------
__global__ __launch_bounds__(256) void ops_prep_kernel(
    const float* __restrict__ node,     // [768,384]
    const float* __restrict__ gamma,    // [384]
    const float* __restrict__ beta,     // [384]
    const float* __restrict__ W_embed,  // [64,384]
    const float* __restrict__ b_embed,  // [64]
    const float* __restrict__ W_final,  // [128,1024]
    __bf16* __restrict__ b_out,         // [768,32]
    __bf16* __restrict__ tmp_out)       // [768,128*32]
{
    __shared__ float xr[NODE_DIM];
    __shared__ float red[512];
    __shared__ float part[256];
    __shared__ float af[INNER];

    const int i = blockIdx.x;
    const int t = threadIdx.x;

    // ---- load row + running sums for mean/var ----
    float s = 0.f, s2 = 0.f;
    for (int d = t; d < NODE_DIM; d += 256) {
        float v = node[(size_t)i * NODE_DIM + d];
        xr[d] = v;
        s += v;
        s2 += v * v;
    }
    red[t] = s;
    red[256 + t] = s2;
    __syncthreads();
    for (int off = 128; off > 0; off >>= 1) {
        if (t < off) {
            red[t] += red[t + off];
            red[256 + t] += red[256 + t + off];
        }
        __syncthreads();
    }
    const float mu   = red[0] * (1.0f / NODE_DIM);
    const float var  = red[256] * (1.0f / NODE_DIM) - mu * mu;
    const float rstd = rsqrtf(var + LN_EPS);

    // ---- normalize in place (each thread touches only its own elements) ----
    for (int d = t; d < NODE_DIM; d += 256)
        xr[d] = (xr[d] - mu) * rstd * gamma[d] + beta[d];
    __syncthreads();

    // ---- h[m] = xr . W_embed[m,:] + b_embed[m]  (4 partials per output) ----
    {
        const int m = t & 63;
        const int q = t >> 6;                 // quarter of the 384-dim dot
        const float* wr = W_embed + (size_t)m * NODE_DIM + q * 96;
        const float* xp = xr + q * 96;
        float p = 0.f;
        #pragma unroll 4
        for (int d = 0; d < 96; ++d) p += xp[d] * wr[d];
        part[t] = p;
    }
    __syncthreads();
    if (t < 64) {
        float h = part[t] + part[64 + t] + part[128 + t] + part[192 + t] + b_embed[t];
        if (t < INNER) af[t] = h;                                    // a-half
        else           b_out[(size_t)i * INNER + (t - INNER)] = (__bf16)h; // b-half
    }
    __syncthreads();

    // ---- tmp[e,k] = sum_d af[d] * W_final[e, d*32+k]; 16 outputs/thread ----
    {
        const int base = t * 16;          // flat index into 128*32
        const int e    = base >> 5;
        const int k0   = base & 31;       // 0 or 16
        float acc[16];
        #pragma unroll
        for (int q = 0; q < 16; ++q) acc[q] = 0.f;
        const float* wbase = W_final + (size_t)e * (INNER * INNER) + k0;
        #pragma unroll 4
        for (int d = 0; d < INNER; ++d) {
            const float av = af[d];
            const float4* w4 = (const float4*)(wbase + d * INNER);
            #pragma unroll
            for (int q = 0; q < 4; ++q) {
                float4 w = w4[q];
                acc[4 * q + 0] += av * w.x;
                acc[4 * q + 1] += av * w.y;
                acc[4 * q + 2] += av * w.z;
                acc[4 * q + 3] += av * w.w;
            }
        }
        __bf16* dst = tmp_out + (size_t)i * (EDGE_DIM * INNER) + base;
        #pragma unroll
        for (int q = 0; q < 16; ++q) dst[q] = (__bf16)acc[q];
    }
}

// ---------------------------------------------------------------------------
// Kernel 2: main GEMM + epilogue.
//   grid = (6 j-chunks, 768 i), block = 256 (8 waves).
//   Per block: out[i, jc*128 .. +128, 0..128]
//     = b_chunk[128,32] @ tmp_i[128,32]^T + edge + b_final
//   WMMA orientation: M = e (tile of 16), N = j (16 per wave), K = 32.
//   LDS rows padded to 40 bf16 (80 B) to avoid bank conflicts.
// ---------------------------------------------------------------------------
#define LDS_PITCH 40

__global__ __launch_bounds__(256) void ops_main_kernel(
    const float*  __restrict__ edge,     // [768,768,128]
    const float*  __restrict__ b_final,  // [128]
    const __bf16* __restrict__ bmat,     // [768,32]
    const __bf16* __restrict__ tmp,      // [768,128*32]
    float* __restrict__ out)             // [768,768,128]
{
    __shared__ __bf16 tmp_lds[EDGE_DIM * LDS_PITCH]; // 128 rows (e), 32 used cols (k)
    __shared__ __bf16 b_lds[128 * LDS_PITCH];        // 128 rows (j), 32 used cols (k)
    __shared__ float  bf_lds[EDGE_DIM];

    const int i   = blockIdx.y;
    const int jc  = blockIdx.x;         // 0..5 (chunks of 128 rows of j)
    const int tid = threadIdx.x;

    if (tid < EDGE_DIM) bf_lds[tid] = b_final[tid];

    // stage operands: each thread copies one 32 B half-row (16 bf16)
    {
        const int r = tid >> 1;
        const int h = tid & 1;
        const uint4* gs = (const uint4*)(tmp + (size_t)i * (EDGE_DIM * INNER) + r * INNER + h * 16);
        uint4* ld = (uint4*)(&tmp_lds[r * LDS_PITCH + h * 16]);
        ld[0] = gs[0];
        ld[1] = gs[1];
        const uint4* gb = (const uint4*)(bmat + (size_t)(jc * 128 + r) * INNER + h * 16);
        uint4* lb = (uint4*)(&b_lds[r * LDS_PITCH + h * 16]);
        lb[0] = gb[0];
        lb[1] = gb[1];
    }
    __syncthreads();

    const int wave = tid >> 5;
    const int lane = tid & 31;
    const int n    = lane & 15;   // N index (j within tile) / M row for A reads
    const int half = lane >> 4;

    // B operand (32x16, K x N): lane holds N = n, K = half*16 .. half*16+15
    // -> one contiguous 32 B read from b row (wave*16 + n).
    v16bf B;
    {
        const uint4* p = (const uint4*)(&b_lds[(wave * 16 + n) * LDS_PITCH + half * 16]);
        ((uint4*)&B)[0] = p[0];
        ((uint4*)&B)[1] = p[1];
    }

    const int j = jc * 128 + wave * 16 + n;
    const size_t rowbase = ((size_t)i * N_NODES + j) * EDGE_DIM;
    const float* erow = edge + rowbase;
    float*       orow = out + rowbase;

    #pragma unroll
    for (int t = 0; t < 8; ++t) {       // 8 e-tiles of 16
        // A operand (16x32, M x K): lane holds M = n,
        // K = half*16 .. half*16+15 -> two 16 B reads.
        v16bf A;
        {
            const __bf16* arow = &tmp_lds[(t * 16 + n) * LDS_PITCH];
            ((uint4*)&A)[0] = *(const uint4*)(arow + half * 16);
            ((uint4*)&A)[1] = *(const uint4*)(arow + half * 16 + 8);
        }
        v8f acc = {};
        acc = __builtin_amdgcn_wmma_f32_16x16x32_bf16(
            /*neg_a=*/false, A, /*neg_b=*/false, B,
            /*c_mod=*/(short)0, acc, /*reuse_a=*/false, /*reuse_b=*/false);

        // C/D layout: reg r <-> e = t*16 + half*8 + r  (8 consecutive e/lane)
        const int e0 = t * 16 + half * 8;
        float4 ea = *(const float4*)(erow + e0);
        float4 eb = *(const float4*)(erow + e0 + 4);
        float4 fa = *(const float4*)(&bf_lds[e0]);
        float4 fb = *(const float4*)(&bf_lds[e0 + 4]);
        float4 o0, o1;
        o0.x = acc[0] + ea.x + fa.x;
        o0.y = acc[1] + ea.y + fa.y;
        o0.z = acc[2] + ea.z + fa.z;
        o0.w = acc[3] + ea.w + fa.w;
        o1.x = acc[4] + eb.x + fb.x;
        o1.y = acc[5] + eb.y + fb.y;
        o1.z = acc[6] + eb.z + fb.z;
        o1.w = acc[7] + eb.w + fb.w;
        *(float4*)(orow + e0)     = o0;
        *(float4*)(orow + e0 + 4) = o1;
    }
}

// ---------------------------------------------------------------------------
// Launch
// ---------------------------------------------------------------------------
extern "C" void kernel_launch(void* const* d_in, const int* in_sizes, int n_in,
                              void* d_out, int out_size, void* d_ws, size_t ws_size,
                              hipStream_t stream) {
    const float* node    = (const float*)d_in[0];
    const float* edge    = (const float*)d_in[1];
    const float* gamma   = (const float*)d_in[2];
    const float* beta    = (const float*)d_in[3];
    const float* W_embed = (const float*)d_in[4];
    const float* b_embed = (const float*)d_in[5];
    const float* W_final = (const float*)d_in[6];
    const float* b_final = (const float*)d_in[7];
    float* out = (float*)d_out;

    // workspace: b_bf16 [768*32] then tmp_bf16 [768*128*32]  (~6.34 MB)
    __bf16* b_ws   = (__bf16*)d_ws;
    __bf16* tmp_ws = (__bf16*)((char*)d_ws + (size_t)N_NODES * INNER * sizeof(__bf16));

    ops_prep_kernel<<<N_NODES, 256, 0, stream>>>(
        node, gamma, beta, W_embed, b_embed, W_final, b_ws, tmp_ws);

    dim3 grid(N_NODES / 128, N_NODES);   // (6 j-chunks, 768 i)
    ops_main_kernel<<<grid, 256, 0, stream>>>(edge, b_final, b_ws, tmp_ws, out);
}
